// LstmModel2_71777493450905
// MI455X (gfx1250) — matrix operand order repaired and checked by reading, not compiled, not run on hardware
//
#include <hip/hip_runtime.h>
#include <hip/hip_bf16.h>
#include <cstdint>

// ---------------- CDNA5 vector types ----------------
typedef __attribute__((ext_vector_type(16))) __bf16 v16bf;
typedef __attribute__((ext_vector_type(8)))  float  v8f;
typedef __attribute__((ext_vector_type(8)))  unsigned short us8;
typedef __attribute__((ext_vector_type(4)))  float  f4;

#define B_SZ  64
#define T_LEN 128
#define S_N   100
#define A_DIM 16
#define H_DIM 120
#define ROWS  (B_SZ * S_N)     // 6400
#define TAU_F 5.0f

// padded GEMM geometry: N = 4 gates x 128 cols, K = [A(16) | H(120->128) | pad]
#define ND 512
#define KD 152                 // stored K stride in halves (bank-conflict-free); K>=144 logical zeros
#define HROW 136               // h LDS row stride in halves (conflict-free)
#define SMEM_W    (ND * KD * 2)        // 155648 B
#define SMEM_BIAS (ND * 4)             // 2048 B
#define SMEM_H    (16 * HROW * 2)      // 4352 B
#define SMEM_TOTAL (SMEM_W + SMEM_BIAS + SMEM_H)   // 162048 B

union Frag {
  v16bf bf;
  us8 u8[2];
  unsigned short u[16];
};

__device__ __forceinline__ unsigned short f2bf(float x) {
  unsigned int u = __float_as_uint(x);
  u += 0x7FFFu + ((u >> 16) & 1u);   // round-to-nearest-even
  return (unsigned short)(u >> 16);
}
__device__ __forceinline__ float sigm(float x) { return 1.0f / (1.0f + __expf(-x)); }
__device__ __forceinline__ float tanh_fast(float x) {
  x = fminf(fmaxf(x, -15.0f), 15.0f);
  float e = __expf(2.0f * x);
  return (e - 1.0f) / (e + 1.0f);
}

// ---------------- Phase 1: batched LSTM via bf16 WMMA ----------------
// One 16-row M-tile per workgroup (400 WGs). 4 waves, each owns 2 gate quadruples
// (tiles q, q+8, q+16, q+24 = i,f,g,o for h-cols [16q,16q+16)).
__global__ void __launch_bounds__(128)
lstm_wmma_kernel(const float* __restrict__ x, const float* __restrict__ W_ih,
                 const float* __restrict__ W_hh, const float* __restrict__ b_ih,
                 const float* __restrict__ b_hh, float* __restrict__ hout)
{
  extern __shared__ unsigned char smem[];
  unsigned short* Wt   = (unsigned short*)smem;                        // [ND][KD] bf16, B-matrix (k-major per col)
  float*          bias = (float*)(smem + SMEM_W);                      // [ND]
  unsigned short* hb   = (unsigned short*)(smem + SMEM_W + SMEM_BIAS); // [16][HROW] bf16 h state

  const int tid  = threadIdx.x;
  const int wave = tid >> 5;
  const int lane = tid & 31;
  const int l15  = lane & 15;
  const int hlf  = lane >> 4;
  const int rowbase = blockIdx.x * 16;

  // ---- cooperative weight prep: Wt[n][k] = B[k][n] (bf16), gates padded 120->128 ----
  for (int idx = tid; idx < ND * KD; idx += 128) {
    int n = idx / KD, k = idx - n * KD;
    int gate = n >> 7, col = n & 127;
    float v = 0.0f;
    if (col < H_DIM) {
      int orow = gate * H_DIM + col;                 // row of (4H x *) weight
      if (k < A_DIM)              v = W_ih[orow * A_DIM + k];
      else if (k < A_DIM + H_DIM) v = W_hh[orow * H_DIM + (k - A_DIM)];
    }
    Wt[n * KD + k] = f2bf(v);
  }
  for (int n = tid; n < ND; n += 128) {
    int gate = n >> 7, col = n & 127;
    bias[n] = (col < H_DIM) ? (b_ih[gate * H_DIM + col] + b_hh[gate * H_DIM + col]) : 0.0f;
  }
  for (int i = tid; i < 16 * HROW; i += 128) hb[i] = 0;
  __syncthreads();

  // per-lane x stream pointer: row = rowbase + l15, x[b][t][s][k0..k0+7], k0 = 8*hlf
  const int row = rowbase + l15;
  const int bb = row / S_N;
  const int ss = row - bb * S_N;
  const float* xptr = x + ((size_t)bb * T_LEN * S_N + ss) * A_DIM + hlf * 8;

  v8f cst[2];                                        // c-state: 2 quadruple tiles, C/D layout
  #pragma unroll
  for (int qq = 0; qq < 2; ++qq)
    #pragma unroll
    for (int j = 0; j < 8; ++j) cst[qq][j] = 0.0f;

  const us8 zero8 = {0, 0, 0, 0, 0, 0, 0, 0};

  for (int t = 0; t < T_LEN; ++t) {
    // ---- build A fragments (16-bit A layout: lane=M, e<8 -> K=8*hlf+e, e>=8 -> K=16+8*hlf+e-8) ----
    Frag fa[5];
    {
      f4 x0 = *(const f4*)(xptr);
      f4 x1 = *(const f4*)(xptr + 4);
      xptr += S_N * A_DIM;
      fa[0].u[0] = f2bf(x0.x); fa[0].u[1] = f2bf(x0.y); fa[0].u[2] = f2bf(x0.z); fa[0].u[3] = f2bf(x0.w);
      fa[0].u[4] = f2bf(x1.x); fa[0].u[5] = f2bf(x1.y); fa[0].u[6] = f2bf(x1.z); fa[0].u[7] = f2bf(x1.w);
      const unsigned short* hr = hb + l15 * HROW;
      fa[0].u8[1] = *(const us8*)(hr + 8 * hlf);                         // h cols 0..15 region
      #pragma unroll
      for (int c = 1; c < 4; ++c) {
        fa[c].u8[0] = *(const us8*)(hr + (32 * c - 16 + 8 * hlf));
        fa[c].u8[1] = *(const us8*)(hr + (32 * c + 8 * hlf));
      }
      fa[4].u8[0] = *(const us8*)(hr + (112 + 8 * hlf));                 // h cols 112..127
      fa[4].u8[1] = zero8;                                               // K 144..159 pad
    }
    __syncthreads();   // all waves done reading h_{t-1}

    #pragma unroll
    for (int qq = 0; qq < 2; ++qq) {
      const int q = 2 * wave + qq;
      v8f acc[4];
      #pragma unroll
      for (int g = 0; g < 4; ++g) {
        float bv = bias[(q + 8 * g) * 16 + l15];
        #pragma unroll
        for (int j = 0; j < 8; ++j) acc[g][j] = bv;
      }
      #pragma unroll
      for (int c = 0; c < 5; ++c) {
        #pragma unroll
        for (int g = 0; g < 4; ++g) {
          const int nt = q + 8 * g;
          const unsigned short* wp = Wt + (nt * 16 + l15) * KD + 32 * c + 8 * hlf;
          Frag wb;
          if (c < 4) {
            wb.u8[0] = *(const us8*)(wp);
            wb.u8[1] = *(const us8*)(wp + 16);
          } else {
            wb.u8[0] = *(const us8*)(wp);                 // K 128+8*hlf..+7
            wb.u8[1] = hlf ? zero8 : *(const us8*)(wp + 16);  // hlf=0: stored zeros 144..151; K>=152 logical zeros
          }
          acc[g] = __builtin_amdgcn_wmma_f32_16x16x32_bf16(
              false, fa[c].bf, false, wb.bf, (short)0, acc[g], false, false);
        }
      }
      // ---- gate combine: c = f*c + i*g ; h = o*tanh(c) (C/D layout: M = j + 8*hlf, N = l15) ----
      #pragma unroll
      for (int j = 0; j < 8; ++j) {
        float iv = sigm(acc[0][j]);
        float fv = sigm(acc[1][j]);
        float gv = tanh_fast(acc[2][j]);
        float ov = sigm(acc[3][j]);
        float cv = fv * cst[qq][j] + iv * gv;
        cst[qq][j] = cv;
        float hv = ov * tanh_fast(cv);
        int hrow_i = j + 8 * hlf;
        hb[hrow_i * HROW + q * 16 + l15] = f2bf(hv);
        if (t == T_LEN - 1) {
          int hcol = q * 16 + l15;
          if (hcol < H_DIM)
            hout[(size_t)(rowbase + hrow_i) * H_DIM + hcol] = hv;
        }
      }
    }
    __syncthreads();   // h_t fully written before next read
  }
}

// ---------------- Phase 2: score linear (H -> 1) ----------------
__global__ void score_kernel(const float* __restrict__ h, const float* __restrict__ w_lin,
                             const float* __restrict__ b_lin, float* __restrict__ s)
{
  int r = blockIdx.x * blockDim.x + threadIdx.x;
  if (r >= ROWS) return;
  const float* hp = h + (size_t)r * H_DIM;
  float acc = b_lin[0];
  #pragma unroll 4
  for (int k = 0; k < H_DIM; ++k) acc += hp[k] * w_lin[k];
  s[r] = acc;
}

// ---------------- Phase 3: NeuralSort relaxed permutation ----------------
// p_hat[b,i,j] = softmax_j( (s_j * (S-1-2i) - rowsum_j) / TAU )
__global__ void __launch_bounds__(128)
sort_kernel(const float* __restrict__ s, float* __restrict__ p)
{
  __shared__ float sv[S_N];
  __shared__ float rs[S_N];
  const int b = blockIdx.x, tid = threadIdx.x;
  if (tid < S_N) sv[tid] = s[b * S_N + tid];
  __syncthreads();
  if (tid < S_N) {
    float si = sv[tid], a = 0.0f;
    for (int j = 0; j < S_N; ++j) a += fabsf(si - sv[j]);
    rs[tid] = a;
  }
  __syncthreads();
  for (int i = tid; i < S_N; i += blockDim.x) {
    float scal = (float)(S_N - 1 - 2 * i);
    float m = -3.0e38f;
    for (int j = 0; j < S_N; ++j) m = fmaxf(m, sv[j] * scal - rs[j]);
    float sum = 0.0f;
    float* outp = p + ((size_t)b * S_N + i) * S_N;
    for (int j = 0; j < S_N; ++j) {
      float e = __expf(((sv[j] * scal - rs[j]) - m) * (1.0f / TAU_F));
      outp[j] = e;
      sum += e;
    }
    float inv = 1.0f / sum;
    for (int j = 0; j < S_N; ++j) outp[j] *= inv;
  }
}

// ---------------- Phase 4: MLP 10000->256 (relu) ->100, softmax ----------------
__global__ void __launch_bounds__(256)
mlp_kernel(const float* __restrict__ p, const float* __restrict__ W2, const float* __restrict__ b2,
           const float* __restrict__ W3, const float* __restrict__ b3, float* __restrict__ out)
{
  __shared__ float z1[256];
  __shared__ float z2[S_N];
  const int b = blockIdx.x, n = threadIdx.x;
  const float* pb = p + (size_t)b * (S_N * S_N);
  const float* w  = W2 + (size_t)n * (S_N * S_N);
  float acc = b2[n];
  for (int j = 0; j < S_N * S_N; j += 4) {
    f4 pv = *(const f4*)(pb + j);
    f4 wv = *(const f4*)(w + j);
    acc += pv.x * wv.x + pv.y * wv.y + pv.z * wv.z + pv.w * wv.w;
  }
  z1[n] = fmaxf(acc, 0.0f);
  __syncthreads();
  if (n < S_N) {
    float a2 = b3[n];
    const float* w3 = W3 + n * 256;
    #pragma unroll 4
    for (int k = 0; k < 256; ++k) a2 += z1[k] * w3[k];
    z2[n] = a2;
  }
  __syncthreads();
  if (n < S_N) {
    float m = -3.0e38f;
    for (int k = 0; k < S_N; ++k) m = fmaxf(m, z2[k]);
    float sum = 0.0f;
    for (int k = 0; k < S_N; ++k) sum += __expf(z2[k] - m);
    out[b * S_N + n] = __expf(z2[n] - m) / sum;
  }
}

// ---------------- launch ----------------
extern "C" void kernel_launch(void* const* d_in, const int* in_sizes, int n_in,
                              void* d_out, int out_size, void* d_ws, size_t ws_size,
                              hipStream_t stream)
{
  (void)in_sizes; (void)n_in; (void)out_size; (void)ws_size;
  const float* x     = (const float*)d_in[0];
  const float* W_ih  = (const float*)d_in[1];
  const float* W_hh  = (const float*)d_in[2];
  const float* b_ih  = (const float*)d_in[3];
  const float* b_hh  = (const float*)d_in[4];
  const float* w_lin = (const float*)d_in[5];
  const float* b_lin = (const float*)d_in[6];
  const float* W2    = (const float*)d_in[7];
  const float* b2    = (const float*)d_in[8];
  const float* W3    = (const float*)d_in[9];
  const float* b3    = (const float*)d_in[10];

  float* ws   = (float*)d_ws;
  float* hbuf = ws;                                   // 6400*120
  float* sbuf = ws + (size_t)ROWS * H_DIM;            // 6400
  float* pbuf = sbuf + ROWS;                          // 64*10000 (16B aligned)

  static_assert(SMEM_TOTAL <= 160 * 1024, "LDS budget");
  hipFuncSetAttribute((const void*)lstm_wmma_kernel,
                      hipFuncAttributeMaxDynamicSharedMemorySize, SMEM_TOTAL);

  lstm_wmma_kernel<<<ROWS / 16, 128, SMEM_TOTAL, stream>>>(x, W_ih, W_hh, b_ih, b_hh, hbuf);
  score_kernel<<<(ROWS + 255) / 256, 256, 0, stream>>>(hbuf, w_lin, b_lin, sbuf);
  sort_kernel<<<B_SZ, 128, 0, stream>>>(sbuf, pbuf);
  mlp_kernel<<<B_SZ, 256, 0, stream>>>(pbuf, W2, b2, W3, b3, (float*)d_out);
}
